// BasicRNN_65300682769181
// MI455X (gfx1250) — matrix operand order repaired and checked by reading, compiled, not run on hardware
//
#include <hip/hip_runtime.h>
#include <hip/hip_bf16.h>
#include <math.h>

typedef __attribute__((ext_vector_type(16))) _Float16 v16h;
typedef __attribute__((ext_vector_type(8)))  _Float16 v8h;
typedef __attribute__((ext_vector_type(8)))  float    v8f;

#define RNN_B   64
#define RNN_T   512
#define RNN_NIN 256
#define RNN_N   512
#define NWG     32          // one WG per 16 output columns

// workspace layout (bytes)
#define WS_COUNTER 0
#define WS_HBUF    1024                          // 2 * 64*512 f16 = 128KB
#define WS_WIH     (WS_HBUF + 2*RNN_B*RNN_N*2)   // 512*256 f16 = 256KB
#define WS_WHH     (WS_WIH + RNN_N*RNN_NIN*2)    // 512*512 f16 = 512KB

// ---------------------------------------------------------------- converts
__global__ void rnn_cvt_weights(const float* __restrict__ wih,
                                const float* __restrict__ whh,
                                _Float16* __restrict__ wih16,
                                _Float16* __restrict__ whh16,
                                int* __restrict__ counter) {
    int i = blockIdx.x * blockDim.x + threadIdx.x;
    if (i == 0) *counter = 0;               // reset grid barrier every launch
    if (i < RNN_N * RNN_NIN) wih16[i] = (_Float16)wih[i];
    if (i < RNN_N * RNN_N)   whh16[i] = (_Float16)whh[i];
}

// ---------------------------------------------------------------- xp = x @ W_ih^T  -> out[b][t][n]
__global__ __launch_bounds__(256) void rnn_proj(const float* __restrict__ x,
                                                const _Float16* __restrict__ wih16,
                                                float* __restrict__ out) {
    const int lane  = threadIdx.x & 31;
    const int wid   = blockIdx.x * (blockDim.x >> 5) + (threadIdx.x >> 5);
    const int rt    = wid >> 5;              // 2048 row tiles (M = B*T = 32768)
    const int nt    = wid & 31;              // 32 col tiles  (N = 512)
    const int r0    = rt * 16, n0 = nt * 16;
    const int l15   = lane & 15;
    const int khalf = lane >> 4;
    const int row   = r0 + l15;              // A fragment row (per lane)
    const int col   = n0 + l15;              // B fragment column (per lane)

    v8f c = {};
#pragma unroll
    for (int kt = 0; kt < RNN_NIN; kt += 32) {
        // A: x is fp32 row-major [32768][256]; convert to f16 in-register
        const float* pa = x + row * RNN_NIN + kt + khalf * 16;
        v16h a;
#pragma unroll
        for (int i = 0; i < 8; ++i) { a[i] = (_Float16)pa[i]; a[8 + i] = (_Float16)pa[8 + i]; }
        // B: B[k][n] = W_ih[n][k]; 16 contiguous f16 per lane
        v16h b = *(const v16h*)(wih16 + col * RNN_NIN + kt + khalf * 16);
        c = __builtin_amdgcn_wmma_f32_16x16x32_f16(false, a, false, b, (short)0, c, false, false);
    }
    // C/D layout: lane half selects M offset 0/8, 8 rows per lane
    float* po = out + (r0 + khalf * 8) * RNN_N + n0 + l15;
#pragma unroll
    for (int j = 0; j < 8; ++j) po[j * RNN_N] = c[j];
}

// ---------------------------------------------------------------- grid barrier (monotonic, device scope)
__device__ __forceinline__ void grid_barrier(int* counter, int target) {
    __threadfence();                 // release all prior global stores to device scope
    __syncthreads();
    if (threadIdx.x == 0) {
        __hip_atomic_fetch_add(counter, 1, __ATOMIC_RELAXED, __HIP_MEMORY_SCOPE_AGENT);
        while (__hip_atomic_load(counter, __ATOMIC_RELAXED, __HIP_MEMORY_SCOPE_AGENT) < target)
            __builtin_amdgcn_s_sleep(1);
    }
    __syncthreads();
    __threadfence();                 // acquire: invalidate stale lines
}

// ---------------------------------------------------------------- serial recurrence, persistent kernel
__global__ __launch_bounds__(128) void rnn_recur(float* __restrict__ out,
                                                 const _Float16* __restrict__ whh16,
                                                 _Float16* __restrict__ hbuf,
                                                 int* __restrict__ counter) {
    const int lane  = threadIdx.x & 31;
    const int wv    = threadIdx.x >> 5;      // 4 waves: 16 batch rows each
    const int n0    = blockIdx.x * 16;       // this WG's output columns
    const int m0    = wv * 16;
    const int l15   = lane & 15;
    const int khalf = lane >> 4;
    const int col   = n0 + l15;
    const int rbase = m0 + khalf * 8;        // C/D row base for this lane

    // Preload all 16 B fragments of W_hh^T for our 16 columns into VGPRs.
    // B[k][n] = W_hh[n][k] -> 16 contiguous f16 per lane per fragment.
    v16h bf[16];
#pragma unroll
    for (int kt = 0; kt < 16; ++kt)
        bf[kt] = *(const v16h*)(whh16 + col * RNN_N + kt * 32 + khalf * 16);

    _Float16* h0 = hbuf;                     // double-buffered tanh(h) in f16
    _Float16* h1 = hbuf + RNN_B * RNN_N;

    // t=0: h_0 = 0 so out[:,0,:] = xp_0 (already there). Seed h1 = tanh(xp_0).
    {
        float* po = out + rbase * (RNN_T * RNN_N) + col;   // t = 0
#pragma unroll
        for (int j = 0; j < 8; ++j) {
            float v = po[j * (RNN_T * RNN_N)];
            h1[(rbase + j) * RNN_N + col] = (_Float16)tanhf(v);
        }
    }
    grid_barrier(counter, NWG);

    for (int t = 1; t < RNN_T; ++t) {
        const _Float16* hp = (t & 1) ? h1 : h0;
        _Float16*       hn = (t & 1) ? h0 : h1;

        v8f c = {};
#pragma unroll
        for (int kt = 0; kt < 16; ++kt) {
            const _Float16* pa = hp + (m0 + l15) * RNN_N + kt * 32 + khalf * 16;
            v8h lo = *(const v8h*)pa;         // K = kt*32 + khalf*16 .. +7
            v8h hi = *(const v8h*)(pa + 8);   // K = kt*32 + khalf*16 + 8 .. +15
            v16h a;
#pragma unroll
            for (int i = 0; i < 8; ++i) { a[i] = lo[i]; a[8 + i] = hi[i]; }
            c = __builtin_amdgcn_wmma_f32_16x16x32_f16(false, a, false, bf[kt], (short)0, c, false, false);
        }

        // h_t = (tanh(h_{t-1}) @ W_hh^T) + xp_t ; store pre-activation, publish tanh(h_t)
        float* po = out + rbase * (RNN_T * RNN_N) + t * RNN_N + col;
#pragma unroll
        for (int j = 0; j < 8; ++j) {
            float v = c[j] + po[j * (RNN_T * RNN_N)];
            po[j * (RNN_T * RNN_N)] = v;
            hn[(rbase + j) * RNN_N + col] = (_Float16)tanhf(v);
        }
        grid_barrier(counter, (t + 1) * NWG);
    }
}

// ---------------------------------------------------------------- launcher
extern "C" void kernel_launch(void* const* d_in, const int* in_sizes, int n_in,
                              void* d_out, int out_size, void* d_ws, size_t ws_size,
                              hipStream_t stream) {
    const float* x   = (const float*)d_in[0];   // [64,512,256]
    const float* wih = (const float*)d_in[1];   // [512,256]
    const float* whh = (const float*)d_in[2];   // [512,512]
    float* out = (float*)d_out;                 // [64,512,512]

    char* ws = (char*)d_ws;
    int*      counter = (int*)     (ws + WS_COUNTER);
    _Float16* hbuf    = (_Float16*)(ws + WS_HBUF);
    _Float16* wih16   = (_Float16*)(ws + WS_WIH);
    _Float16* whh16   = (_Float16*)(ws + WS_WHH);

    rnn_cvt_weights<<<1024, 256, 0, stream>>>(wih, whh, wih16, whh16, counter);
    // 2048 row-tiles * 32 col-tiles = 65536 waves, 8 waves per 256-thread block
    rnn_proj<<<8192, 256, 0, stream>>>(x, wih16, out);
    rnn_recur<<<NWG, 128, 0, stream>>>(out, whh16, hbuf, counter);
}
